// Intervolution_83863531421942
// MI455X (gfx1250) — compile-verified, matrix operand order checked by reading, not run on hardware
//
#include <hip/hip_runtime.h>

typedef __attribute__((ext_vector_type(16))) __bf16 v16bf;
typedef __attribute__((ext_vector_type(8)))  float  v8f;

#define B_   8
#define H_   56
#define W_   56
#define C_   192
#define NH_  6
#define HD_  32
#define AC_  486
#define NPIX (B_ * H_ * W_)          /* 25088 */
#define SCALE_ 0.17677669529663689f  /* 32^-0.5 */
#define EPS_ 1e-5f

// Load a 16x32 bf16 operand tile from row-major f32 memory, in the CDNA5
// 16-bit A-matrix VGPR layout (ISA 7.12.2):
//   lanes 0-15 : row=lane,    K = {0..7} then {16..23}
//   lanes 16-31: row=lane-16, K = {8..15} then {24..31}
// (w[o][c] is B^T row-major, so the same loader serves the B operand.)
// All source addresses are 32B-aligned (rows = 192 f32, k-offsets = multiples
// of 8 f32) so the float4 loads lower to global_load_b128.
__device__ __forceinline__ v16bf load_tile_16x32(const float* __restrict__ base,
                                                 int ld, int rmax) {
  const int lane = threadIdx.x & 31;
  int r = lane & 15;
  if (r > rmax) r = rmax;                 // clamp for N not multiple of 16
  const int kofs = (lane & 16) ? 8 : 0;
  const float* p = base + (size_t)r * ld + kofs;
  const float4 x0 = reinterpret_cast<const float4*>(p)[0];       // K+0..3
  const float4 x1 = reinterpret_cast<const float4*>(p)[1];       // K+4..7
  const float4 x2 = reinterpret_cast<const float4*>(p + 16)[0];  // K+16..19
  const float4 x3 = reinterpret_cast<const float4*>(p + 16)[1];  // K+20..23
  v16bf t;
  t[0]  = (__bf16)x0.x; t[1]  = (__bf16)x0.y; t[2]  = (__bf16)x0.z; t[3]  = (__bf16)x0.w;
  t[4]  = (__bf16)x1.x; t[5]  = (__bf16)x1.y; t[6]  = (__bf16)x1.z; t[7]  = (__bf16)x1.w;
  t[8]  = (__bf16)x2.x; t[9]  = (__bf16)x2.y; t[10] = (__bf16)x2.z; t[11] = (__bf16)x2.w;
  t[12] = (__bf16)x3.x; t[13] = (__bf16)x3.y; t[14] = (__bf16)x3.z; t[15] = (__bf16)x3.w;
  return t;
}

__device__ __forceinline__ v8f wmma_bf16(v16bf a, v16bf b, v8f c) {
  return __builtin_amdgcn_wmma_f32_16x16x32_bf16(
      /*neg_a=*/false, a, /*neg_b=*/false, b,
      /*c_mod=*/(short)0, c, /*reuse_a=*/false, /*reuse_b=*/false);
}

// Y[m][n] = act( X[m][:] . Wt[n][:] + bias[n] ),  act = BN+ReLU or identity.
// Register-blocked 2x2: one wave owns a 32M x 32N tile of Y (4 accumulators),
// so each K-step does 4 tile-loads -> 4 WMMAs (2x operand reuse).
// grid = (M/32, ceil(ngroups/8)), block = 256 (8 waves).
template <bool FUSE_BN_RELU>
__global__ __launch_bounds__(256)
void wmma_gemm_kernel(const float* __restrict__ X,    // [M][Kd] row-major
                      const float* __restrict__ Wt,   // [N][Kd] row-major (B^T)
                      const float* __restrict__ bias, // [N]
                      const float* __restrict__ bng,
                      const float* __restrict__ bnb,
                      const float* __restrict__ bnmu,
                      const float* __restrict__ bnvar,
                      float* __restrict__ Y,          // [M][N]
                      int N, int Kd, int ntiles) {
  const int wave    = threadIdx.x >> 5;
  const int ngroups = (ntiles + 1) >> 1;
  const int ng      = blockIdx.y * (blockDim.x >> 5) + wave;
  if (ng >= ngroups) return;              // wave-uniform
  const int m0 = blockIdx.x << 5;
  const int n0 = ng << 5;
  const int rb0 = max(0, N - 1 - n0);     // row clamps for B-operand edge tiles
  const int rb1 = max(0, N - 1 - (n0 + 16));

  v8f acc00 = {}, acc01 = {}, acc10 = {}, acc11 = {};
  for (int k0 = 0; k0 < Kd; k0 += 32) {
    if (k0 + 32 < Kd) {                   // hint next K-step (global_prefetch_b8)
      __builtin_prefetch(X  + (size_t)m0 * Kd + k0 + 32, 0, 1);
      __builtin_prefetch(Wt + (size_t)n0 * Kd + k0 + 32, 0, 1);
    }
    v16bf a0 = load_tile_16x32(X  + (size_t)m0 * Kd + k0,        Kd, 15);
    v16bf a1 = load_tile_16x32(X  + (size_t)(m0 + 16) * Kd + k0, Kd, 15);
    v16bf b0 = load_tile_16x32(Wt + (size_t)n0 * Kd + k0,        Kd, rb0);
    v16bf b1 = load_tile_16x32(Wt + (size_t)(n0 + 16) * Kd + k0, Kd, rb1);
    acc00 = wmma_bf16(a0, b0, acc00);
    acc01 = wmma_bf16(a0, b1, acc01);
    acc10 = wmma_bf16(a1, b0, acc10);
    acc11 = wmma_bf16(a1, b1, acc11);
  }

  // C/D layout: VGPR i -> row mt+i (lanes 0-15) / mt+8+i (lanes 16-31), col = nt + lane%16
  const int lane  = threadIdx.x & 31;
  const int colL  = lane & 15;
  const int rhalf = (lane & 16) ? 8 : 0;
  v8f accs[4] = {acc00, acc01, acc10, acc11};
#pragma unroll
  for (int t = 0; t < 4; ++t) {
    const int col = n0 + ((t & 1) << 4) + colL;
    if (col >= N) continue;               // guards are after all WMMAs
    const int rbase = m0 + ((t >> 1) << 4) + rhalf;
    const float bb = bias[col];
    float scl = 1.f, sft = 0.f;
    if (FUSE_BN_RELU) {
      const float inv = bng[col] * rsqrtf(bnvar[col] + EPS_);
      scl = inv;
      sft = bnb[col] - bnmu[col] * inv;
    }
#pragma unroll
    for (int i = 0; i < 8; ++i) {
      float v = accs[t][i] + bb;
      if (FUSE_BN_RELU) v = fmaxf(v * scl + sft, 0.f);
      Y[(size_t)(rbase + i) * N + col] = v;
    }
  }
}

// softmax over contiguous groups of 9 (layout n*81 + p*9 + q, q fastest), with scale.
__global__ __launch_bounds__(256)
void softmax9_kernel(float* __restrict__ attn, int nrows) {
  const int idx = blockIdx.x * blockDim.x + threadIdx.x;
  if (idx >= nrows) return;
  float* row = attn + (size_t)idx * 9;
  float e[9];
  float m = -3.4e38f;
#pragma unroll
  for (int q = 0; q < 9; ++q) { e[q] = row[q] * SCALE_; m = fmaxf(m, e[q]); }
  float s = 0.f;
#pragma unroll
  for (int q = 0; q < 9; ++q) { e[q] = expf(e[q] - m); s += e[q]; }
  const float inv = 1.f / s;
#pragma unroll
  for (int q = 0; q < 9; ++q) row[q] = e[q] * inv;
}

// Fused "attn x v-patches" + 3x3 scatter-add, algebraically reduced:
//   g[b,y,x,c] = sum_{dy,dx in [-2,2]} W[dy,dx] * v[b, y+dy, x+dx, c]
//   W[dy,dx]   = sum_{(i,j),(qi,qj): qi-i=dy, qj-j=dx, l'=(y+1-i,x+1-j) in range}
//                attn[b, l', head, (i,j), (qi,qj)]
// One block per pixel; one wave per head (lane = head-dim channel). Lanes 0-24
// each build one of the 25 stencil weights; broadcast through LDS.
__global__ __launch_bounds__(192)
void gather_kernel(const float* __restrict__ v,     // [NPIX][C_]
                   const float* __restrict__ attn,  // [NPIX][AC_]
                   float* __restrict__ g) {         // [NPIX][C_]
  __shared__ float Wsh[NH_][32];
  const int pix  = blockIdx.x;
  const int b    = pix / (H_ * W_);
  const int rem  = pix % (H_ * W_);
  const int y    = rem / W_;
  const int x    = rem % W_;
  const int head = threadIdx.x >> 5;
  const int lane = threadIdx.x & 31;

  float wv = 0.f;
  if (lane < 25) {
    const int dy = lane / 5 - 2;
    const int dx = lane % 5 - 2;
#pragma unroll
    for (int i = 0; i < 3; ++i) {
      const int qi = dy + i;
      const int ly = y + 1 - i;
      if (qi < 0 || qi > 2 || ly < 0 || ly >= H_) continue;
#pragma unroll
      for (int j = 0; j < 3; ++j) {
        const int qj = dx + j;
        const int lx = x + 1 - j;
        if (qj < 0 || qj > 2 || lx < 0 || lx >= W_) continue;
        const size_t lp = (size_t)(b * H_ * W_ + ly * W_ + lx);
        wv += attn[lp * AC_ + head * 81 + (i * 3 + j) * 9 + (qi * 3 + qj)];
      }
    }
  }
  Wsh[head][lane] = wv;
  __syncthreads();

  float acc = 0.f;
  const int c = head * HD_ + lane;
#pragma unroll
  for (int k = 0; k < 25; ++k) {
    const int vy = y + k / 5 - 2;
    const int vx = x + k % 5 - 2;
    if (vy < 0 || vy >= H_ || vx < 0 || vx >= W_) continue;
    acc += Wsh[head][k] * v[(size_t)(b * H_ * W_ + vy * W_ + vx) * C_ + c];
  }
  g[(size_t)pix * C_ + c] = acc;
}

extern "C" void kernel_launch(void* const* d_in, const int* in_sizes, int n_in,
                              void* d_out, int out_size, void* d_ws, size_t ws_size,
                              hipStream_t stream) {
  const float* q     = (const float*)d_in[0];
  const float* s     = (const float*)d_in[1];
  const float* v_w   = (const float*)d_in[2];
  const float* v_b   = (const float*)d_in[3];
  const float* v_g   = (const float*)d_in[4];
  const float* v_bt  = (const float*)d_in[5];
  const float* v_mu  = (const float*)d_in[6];
  const float* v_var = (const float*)d_in[7];
  const float* a_w   = (const float*)d_in[8];
  const float* a_b   = (const float*)d_in[9];
  const float* a_g   = (const float*)d_in[10];
  const float* a_bt  = (const float*)d_in[11];
  const float* a_mu  = (const float*)d_in[12];
  const float* a_var = (const float*)d_in[13];
  const float* p_w   = (const float*)d_in[14];
  const float* p_b   = (const float*)d_in[15];
  float* out = (float*)d_out;

  float* vbuf = (float*)d_ws;                    // NPIX * 192 f32 (19.3 MB)
  float* abuf = vbuf + (size_t)NPIX * C_;        // NPIX * 486 f32 (48.8 MB)
  float* gbuf = abuf + (size_t)NPIX * AC_;       // NPIX * 192 f32 (19.3 MB)

  const int mtiles32 = NPIX / 32;                // 784

  // v = relu(BN(q . v_w^T + v_b)) : N=192 -> 12 n-tiles -> 6 groups
  dim3 gv(mtiles32, 1);
  wmma_gemm_kernel<true><<<gv, 256, 0, stream>>>(
      q, v_w, v_b, v_g, v_bt, v_mu, v_var, vbuf, C_, C_, 12);

  // a = relu(BN(s . a_w^T + a_b)) : N=486 -> 31 n-tiles -> 16 groups
  dim3 ga(mtiles32, 2);
  wmma_gemm_kernel<true><<<ga, 256, 0, stream>>>(
      s, a_w, a_b, a_g, a_bt, a_mu, a_var, abuf, AC_, C_, 31);

  // softmax over 9-wide rows
  const int nrows = NPIX * NH_ * 9;              // 1,354,752
  softmax9_kernel<<<(nrows + 255) / 256, 256, 0, stream>>>(abuf, nrows);

  // fused attention-apply + scatter-add (5x5 effective stencil)
  gather_kernel<<<NPIX, 192, 0, stream>>>(vbuf, abuf, gbuf);

  // final projection: out = g . p_w^T + p_b
  wmma_gemm_kernel<false><<<gv, 256, 0, stream>>>(
      gbuf, p_w, p_b, nullptr, nullptr, nullptr, nullptr, out, C_, C_, 12);
}
